// CombinedAttention_57501022158878
// MI455X (gfx1250) — matrix operand order repaired and checked by reading, not compiled, not run on hardware
//
#include <hip/hip_runtime.h>
#include <hip/hip_bf16.h>

// ---------------------------------------------------------------------------
// CDNA5 (gfx1250) combined-attention block, round 3.
// v_wmma_f32_16x16x32_bf16 everywhere; 16x32 wave tiles; K-loop double-buffered.
// ---------------------------------------------------------------------------

typedef __attribute__((ext_vector_type(16))) __bf16         v16bf;
typedef __attribute__((ext_vector_type(8)))  float          v8f;
typedef __attribute__((ext_vector_type(8)))  unsigned short u16x8;

#define HW    4096
#define CCH   240
#define HEADS 8
#define HD    30      // real head dim
#define HDP   32      // padded head dim

union frag16 { unsigned short u[16]; u16x8 h[2]; v16bf v; };

__device__ __forceinline__ unsigned short f2bf(float f) {
    unsigned int u = __builtin_bit_cast(unsigned int, f);
    u += 0x7FFFu + ((u >> 16) & 1u);           // round-to-nearest-even
    return (unsigned short)(u >> 16);
}
__device__ __forceinline__ float bf2f(unsigned short h) {
    unsigned int u = ((unsigned int)h) << 16;
    return __builtin_bit_cast(float, u);
}

__device__ __forceinline__ v8f wmma_bf16(const v16bf& a, const v16bf& b, const v8f& c) {
    return __builtin_amdgcn_wmma_f32_16x16x32_bf16(
        false, a, false, b, (short)0, c, false, false);
}

__device__ __forceinline__ void lds_fence() {
    asm volatile("s_wait_dscnt 0" ::: "memory");
}

// A fragment (16x32, rows = M): lane row m fixed; k = kb+0..15, kb=k0+(lane>=16?16:0)
__device__ __forceinline__ v16bf frag_a(const unsigned short* __restrict__ base,
                                        int stride, int m, int k0, int lane) {
    frag16 f;
    const unsigned short* p = base + (size_t)m * stride + k0 + ((lane >> 4) << 4);
    f.h[0] = *(const u16x8*)(p);
    f.h[1] = *(const u16x8*)(p + 8);
    return f.v;
}
// B fragment (32x16, cols = N): lane col n fixed; k = kb+0..15, kb=k0+(lane>=16?16:0)
// requires column-contiguous storage: element(k,n) at base[n*stride + k]
__device__ __forceinline__ v16bf frag_b(const unsigned short* __restrict__ base,
                                        int stride, int n, int k0, int lane) {
    frag16 f;
    const unsigned short* p = base + (size_t)n * stride + k0 + ((lane >> 4) << 4);
    f.h[0] = *(const u16x8*)(p);
    f.h[1] = *(const u16x8*)(p + 8);
    return f.v;
}

__device__ __forceinline__ float wave_sum(float v) {
#pragma unroll
    for (int off = 16; off > 0; off >>= 1) v += __shfl_down(v, off);
    return v;
}

// ---------------------------------------------------------------------------
// Repack kernels (f32 weights -> padded bf16, zero-filled pads)
// ---------------------------------------------------------------------------
__global__ void repack_w_kernel(const float* __restrict__ in, unsigned short* __restrict__ out,
                                int M, int K, int Kp) {
    int i = blockIdx.x * blockDim.x + threadIdx.x;
    if (i < M * Kp) {
        int m = i / Kp, k = i % Kp;
        out[i] = (k < K) ? f2bf(in[(size_t)m * K + k]) : (unsigned short)0;
    }
}
// OIHW [Cout][Cin][3][3] f32 -> [9][Mp][Cp] bf16
__global__ void repack_w3_kernel(const float* __restrict__ in, unsigned short* __restrict__ out,
                                 int Cout, int Cin, int Mp, int Cp) {
    int i = blockIdx.x * blockDim.x + threadIdx.x;
    int total = 9 * Mp * Cp;
    if (i < total) {
        int tap = i / (Mp * Cp);
        int r = i % (Mp * Cp);
        int m = r / Cp, c = r % Cp;
        unsigned short v = 0;
        if (m < Cout && c < Cin) v = f2bf(in[((size_t)m * Cin + c) * 9 + tap]);
        out[i] = v;
    }
}
// qkv [720][4096] bf16 (d-major) -> QT/KT [8][4096][32] token-major, q pre-scaled d^-0.5
__global__ void repack_qk_kernel(const unsigned short* __restrict__ qkv,
                                 unsigned short* __restrict__ QT,
                                 unsigned short* __restrict__ KT) {
    int i = blockIdx.x * blockDim.x + threadIdx.x;
    if (i < HEADS * HW * HDP) {
        int h = i >> 17;            // / (4096*32)
        int t = (i >> 5) & (HW - 1);
        int d = i & (HDP - 1);
        unsigned short q = 0, k = 0;
        if (d < HD) {
            q = f2bf(bf2f(qkv[(size_t)(h * 90 + d) * HW + t]) * 0.1825741858f); // 30^-0.5
            k = qkv[(size_t)(h * 90 + 30 + d) * HW + t];
        }
        QT[i] = q; KT[i] = k;
    }
}

// ---------------------------------------------------------------------------
// GroupNorm (+optional SiLU): f32 channel-major in -> bf16 channel-last out.
// ---------------------------------------------------------------------------
__global__ void gn_act_kernel(const float* __restrict__ in,
                              const float* __restrict__ gamma,
                              const float* __restrict__ beta,
                              unsigned short* __restrict__ outcl,  // [HW][Cp]
                              int C, int groups, int Cp, int dosilu) {
    const int grp = blockIdx.x;
    const int cpg = C / groups;
    const int n   = cpg * HW;
    const float* base = in + (size_t)grp * n;

    float s = 0.f, ss = 0.f;
    for (int i = threadIdx.x; i < n; i += blockDim.x) {
        float v = base[i];
        s += v; ss += v * v;
    }
    __shared__ float red0[32], red1[32], stat[2];
    const int lane = threadIdx.x & 31, wave = threadIdx.x >> 5;
    float sw = wave_sum(s), ssw = wave_sum(ss);
    if (lane == 0) { red0[wave] = sw; red1[wave] = ssw; }
    __syncthreads();
    if (wave == 0) {
        int nw = blockDim.x >> 5;
        float a = (lane < nw) ? red0[lane] : 0.f;
        float b = (lane < nw) ? red1[lane] : 0.f;
        a = wave_sum(a); b = wave_sum(b);
        if (lane == 0) {
            float mean = a / (float)n;
            float var  = b / (float)n - mean * mean;
            stat[0] = mean;
            stat[1] = rsqrtf(var + 1e-5f);
        }
    }
    __syncthreads();
    const float mean = stat[0], inv = stat[1];
    for (int i = threadIdx.x; i < n; i += blockDim.x) {
        int cl = i >> 12;
        int sp = i & (HW - 1);
        int c  = grp * cpg + cl;
        float v = (base[i] - mean) * inv * gamma[c] + beta[c];
        if (dosilu) v = v / (1.f + __expf(-v));
        outcl[(size_t)sp * Cp + c] = f2bf(v);
    }
}

// ---------------------------------------------------------------------------
// WMMA GEMM: C[M,N] = A[M,Kp] * B(channel-last [N][Kp]).
// Wave tile 16x32 (2 accums share one A fragment); K-loop register
// double-buffered so loads for chunk k+32 are in flight during chunk k's WMMAs.
// block = 256 (8 waves); block tile 16 x 256; grid = (M/16, N/256).
// ---------------------------------------------------------------------------
__global__ void gemm_wmma_kernel(const unsigned short* __restrict__ A,
                                 const unsigned short* __restrict__ Bcl,
                                 const float* __restrict__ bias,
                                 const float* __restrict__ resid,
                                 float* __restrict__ Cf,
                                 unsigned short* __restrict__ Cb,   // bf16 [M][N], optional
                                 int M, int N, int Kp) {
    const int lane = threadIdx.x & 31;
    const int wave = threadIdx.x >> 5;
    const int m0 = blockIdx.x * 16;
    const int n0 = blockIdx.y * 256 + wave * 32;
    const int ma  = m0 + (lane & 15);
    const int nb0 = n0 + (lane & 15);
    const int nb1 = nb0 + 16;

    v8f acc0 = {}, acc1 = {};
    // prologue: first chunk in flight
    v16bf af = frag_a(A,   Kp, ma,  0, lane);
    v16bf b0 = frag_b(Bcl, Kp, nb0, 0, lane);
    v16bf b1 = frag_b(Bcl, Kp, nb1, 0, lane);

    for (int k0 = 0; k0 < Kp; k0 += 32) {
        v16bf afc = af, b0c = b0, b1c = b1;
        const int kn = k0 + 32;
        if (kn < Kp) {                       // issue next chunk before consuming
            af = frag_a(A,   Kp, ma,  kn, lane);
            b0 = frag_b(Bcl, Kp, nb0, kn, lane);
            b1 = frag_b(Bcl, Kp, nb1, kn, lane);
            __builtin_prefetch(Bcl + (size_t)nb0 * Kp + kn + 32, 0, 1);
        }
        acc0 = wmma_bf16(afc, b0c, acc0);
        acc1 = wmma_bf16(afc, b1c, acc1);
    }

    const int mb = m0 + ((lane >> 4) << 3);
#pragma unroll
    for (int t = 0; t < 2; ++t) {
        const int nn = n0 + t * 16 + (lane & 15);
        const v8f& acc = t ? acc1 : acc0;
#pragma unroll
        for (int r = 0; r < 8; ++r) {
            int m = mb + r;
            float v = acc[r];
            if (bias)  v += bias[m];
            if (resid) v += resid[(size_t)m * N + nn];
            if (Cf) Cf[(size_t)m * N + nn] = v;
            if (Cb) Cb[(size_t)m * N + nn] = f2bf(v);
        }
    }
}

// ---------------------------------------------------------------------------
// 3x3 conv (pad=1, 64x64) as implicit GEMM, tap-major K decomposition.
// Wave tile 16x32. W: [9][Mp][Cp]; X: channel-last [HW][Cp]; Y: f32 [Cout][HW].
// ---------------------------------------------------------------------------
__global__ void conv3_wmma_kernel(const unsigned short* __restrict__ W,
                                  const unsigned short* __restrict__ Xcl,
                                  const float* __restrict__ bias,
                                  const float* __restrict__ resid,
                                  float* __restrict__ Yf,
                                  int Cp, int Cout, int Mp) {
    const int lane = threadIdx.x & 31;
    const int wave = threadIdx.x >> 5;
    const int m0 = blockIdx.x * 16;
    const int n0 = blockIdx.y * 256 + wave * 32;
    const int ma = m0 + (lane & 15);

    const int nsp0 = n0 + (lane & 15);
    const int nsp1 = nsp0 + 16;
    const int py0 = nsp0 >> 6, px0 = nsp0 & 63;
    const int py1 = nsp1 >> 6, px1 = nsp1 & 63;

    const u16x8 z8 = {};
    v8f acc0 = {}, acc1 = {};
#pragma unroll
    for (int tap = 0; tap < 9; ++tap) {
        const int dy = tap / 3 - 1, dx = tap % 3 - 1;
        const int y0 = py0 + dy, x0 = px0 + dx;
        const int y1 = py1 + dy, x1 = px1 + dx;
        const bool inb0 = (y0 >= 0) && (y0 < 64) && (x0 >= 0) && (x0 < 64);
        const bool inb1 = (y1 >= 0) && (y1 < 64) && (x1 >= 0) && (x1 < 64);
        const int sp0 = y0 * 64 + x0;
        const int sp1 = y1 * 64 + x1;
        const unsigned short* Wt = W + (size_t)tap * Mp * Cp;
        for (int c0 = 0; c0 < Cp; c0 += 32) {
            v16bf af = frag_a(Wt, Cp, ma, c0, lane);
            frag16 bf0, bf1;
            const int kb = c0 + ((lane >> 4) << 4);
            if (inb0) {
                const unsigned short* p = Xcl + (size_t)sp0 * Cp + kb;
                bf0.h[0] = *(const u16x8*)(p);
                bf0.h[1] = *(const u16x8*)(p + 8);
            } else { bf0.h[0] = z8; bf0.h[1] = z8; }
            if (inb1) {
                const unsigned short* p = Xcl + (size_t)sp1 * Cp + kb;
                bf1.h[0] = *(const u16x8*)(p);
                bf1.h[1] = *(const u16x8*)(p + 8);
            } else { bf1.h[0] = z8; bf1.h[1] = z8; }
            acc0 = wmma_bf16(af, bf0.v, acc0);
            acc1 = wmma_bf16(af, bf1.v, acc1);
        }
    }
    const int mb = m0 + ((lane >> 4) << 3);
#pragma unroll
    for (int t = 0; t < 2; ++t) {
        const int nn = n0 + t * 16 + (lane & 15);
        const v8f& acc = t ? acc1 : acc0;
#pragma unroll
        for (int r = 0; r < 8; ++r) {
            int m = mb + r;
            if (m < Cout) {
                float v = acc[r];
                if (bias)  v += bias[m];
                if (resid) v += resid[(size_t)m * HW + nn];
                Yf[(size_t)m * HW + nn] = v;
            }
        }
    }
}

// ---------------------------------------------------------------------------
// Flash attention: 8 heads, 4096 tokens, head-dim 30 (padded 32).
// QT/KT token-major [h][4096][32] bf16 (Q pre-scaled d^-0.5); V d-major in qkv.
// Output channel-last hsT [4096][256] bf16 (cols 240..255 pre-zeroed).
// ---------------------------------------------------------------------------
__global__ void flash_attn_kernel(const unsigned short* __restrict__ QT,
                                  const unsigned short* __restrict__ KT,
                                  const unsigned short* __restrict__ qkv,
                                  unsigned short* __restrict__ hsT) {
    const int lane = threadIdx.x & 31;
    const int wave = threadIdx.x >> 5;
    const int h  = blockIdx.x;
    const int q0 = blockIdx.y * 128 + wave * 16;

    const unsigned short* Qh = QT + (size_t)h * HW * HDP;
    const unsigned short* Kh = KT + (size_t)h * HW * HDP;
    const unsigned short* V  = qkv + (size_t)(h * 90 + 60) * HW;

    const v16bf qa = frag_a(Qh, HDP, q0 + (lane & 15), 0, lane);

    float rmax[8], rsum[8];
#pragma unroll
    for (int r = 0; r < 8; ++r) { rmax[r] = -1e30f; rsum[r] = 0.f; }
    v8f o0 = {}, o1 = {};

    __shared__ unsigned short pbuf[8][16 * 32];    // per-wave P tile
    unsigned short* P = &pbuf[wave][0];
    const u16x8 z8 = {};

    const int nn = lane & 15;
    const int mb8 = (lane >> 4) << 3;

    for (int t0 = 0; t0 < HW; t0 += 32) {
        v8f s0 = {}, s1 = {};
        {
            v16bf kf0 = frag_b(Kh, HDP, t0 + nn,      0, lane);
            v16bf kf1 = frag_b(Kh, HDP, t0 + 16 + nn, 0, lane);
            s0 = wmma_bf16(qa, kf0, s0);
            s1 = wmma_bf16(qa, kf1, s1);
        }
        float tmax[8];
#pragma unroll
        for (int r = 0; r < 8; ++r) tmax[r] = fmaxf(s0[r], s1[r]);
#pragma unroll
        for (int mk = 1; mk < 16; mk <<= 1)
#pragma unroll
            for (int r = 0; r < 8; ++r)
                tmax[r] = fmaxf(tmax[r], __shfl_xor(tmax[r], mk));

        float corr[8], p0[8], p1[8], tsum[8];
#pragma unroll
        for (int r = 0; r < 8; ++r) {
            float nm = fmaxf(rmax[r], tmax[r]);
            corr[r] = __expf(rmax[r] - nm);
            rmax[r] = nm;
            p0[r] = __expf(s0[r] - nm);
            p1[r] = __expf(s1[r] - nm);
            tsum[r] = p0[r] + p1[r];
        }
#pragma unroll
        for (int mk = 1; mk < 16; mk <<= 1)
#pragma unroll
            for (int r = 0; r < 8; ++r)
                tsum[r] += __shfl_xor(tsum[r], mk);
#pragma unroll
        for (int r = 0; r < 8; ++r) {
            rsum[r] = rsum[r] * corr[r] + tsum[r];
            o0[r] *= corr[r];
            o1[r] *= corr[r];
        }
        // transpose P through LDS (C layout -> A layout)
#pragma unroll
        for (int r = 0; r < 8; ++r) {
            P[(mb8 + r) * 32 + nn]      = f2bf(p0[r]);
            P[(mb8 + r) * 32 + 16 + nn] = f2bf(p1[r]);
        }
        lds_fence();
        frag16 pa;
        {
            const unsigned short* pp = P + (lane & 15) * 32 + ((lane >> 4) << 4);
            pa.h[0] = *(const u16x8*)(pp);
            pa.h[1] = *(const u16x8*)(pp + 8);
        }
        // O += P * V  (V d-major: contiguous along kv per lane)
        {
            frag16 vf;
            const int kb = t0 + ((lane >> 4) << 4);
            {
                const unsigned short* p = V + (size_t)nn * HW + kb;
                vf.h[0] = *(const u16x8*)(p);
                vf.h[1] = *(const u16x8*)(p + 8);
            }
            o0 = wmma_bf16(pa.v, vf.v, o0);
            const int d1 = 16 + nn;
            if (d1 < HD) {
                const unsigned short* p = V + (size_t)d1 * HW + kb;
                vf.h[0] = *(const u16x8*)(p);
                vf.h[1] = *(const u16x8*)(p + 8);
            } else {
                vf.h[0] = z8; vf.h[1] = z8;
            }
            o1 = wmma_bf16(pa.v, vf.v, o1);
        }
        lds_fence();   // protect P before next iteration overwrites it
    }
#pragma unroll
    for (int r = 0; r < 8; ++r) {
        const float inv = 1.f / rsum[r];
        const int m = q0 + mb8 + r;                 // token
        hsT[(size_t)m * 256 + (h * HD + nn)] = f2bf(o0[r] * inv);
        int d1 = 16 + nn;
        if (d1 < HD) hsT[(size_t)m * 256 + (h * HD + d1)] = f2bf(o1[r] * inv);
    }
}

// ---------------------------------------------------------------------------
// SE block + elementwise fusion
// ---------------------------------------------------------------------------
__global__ void pool_kernel(const float* __restrict__ h, float* __restrict__ pooled) {
    const int c = blockIdx.x;
    float s = 0.f;
    for (int i = threadIdx.x; i < HW; i += blockDim.x) s += h[(size_t)c * HW + i];
    __shared__ float red[8];
    const int lane = threadIdx.x & 31, wave = threadIdx.x >> 5;
    float sw = wave_sum(s);
    if (lane == 0) red[wave] = sw;
    __syncthreads();
    if (threadIdx.x == 0) {
        float t = 0.f;
        for (int w = 0; w < 8; ++w) t += red[w];
        pooled[c] = t * (1.f / (float)HW);
    }
}

__global__ void se_kernel(const float* __restrict__ pooled,
                          const float* __restrict__ w1, const float* __restrict__ b1,
                          const float* __restrict__ w2, const float* __restrict__ b2,
                          float* __restrict__ y) {
    __shared__ float pl[CCH];
    __shared__ float mid[8];
    if (threadIdx.x < CCH) pl[threadIdx.x] = pooled[threadIdx.x];
    __syncthreads();
    if (threadIdx.x < 8) {
        float s = b1[threadIdx.x];
        for (int c = 0; c < CCH; ++c) s += w1[threadIdx.x * CCH + c] * pl[c];
        mid[threadIdx.x] = s / (1.f + __expf(-s));          // silu
    }
    __syncthreads();
    if (threadIdx.x < CCH) {
        float s = b2[threadIdx.x];
#pragma unroll
        for (int j = 0; j < 8; ++j) s += w2[threadIdx.x * 8 + j] * mid[j];
        y[threadIdx.x] = 1.f / (1.f + __expf(-s));          // sigmoid
    }
}

__global__ void combine_kernel(const float* __restrict__ h, const float* __restrict__ y,
                               const float* __restrict__ x, const float* __restrict__ sa,
                               float* __restrict__ fuse, int n) {
    int i = blockIdx.x * blockDim.x + threadIdx.x;
    if (i < n) {
        int c = i >> 12;
        fuse[i] = h[i] * y[c] * x[i] + sa[i] + x[i];
    }
}

// ---------------------------------------------------------------------------
// host-side launch
// ---------------------------------------------------------------------------
extern "C" void kernel_launch(void* const* d_in, const int* in_sizes, int n_in,
                              void* d_out, int out_size, void* d_ws, size_t ws_size,
                              hipStream_t stream) {
    const float* x          = (const float*)d_in[0];
    const float* ca_gn1_g   = (const float*)d_in[1];
    const float* ca_gn1_b   = (const float*)d_in[2];
    const float* ca_conv1_w = (const float*)d_in[3];
    const float* ca_conv1_b = (const float*)d_in[4];
    const float* ca_gn2_g   = (const float*)d_in[5];
    const float* ca_gn2_b   = (const float*)d_in[6];
    const float* ca_conv2_w = (const float*)d_in[7];
    const float* ca_conv2_b = (const float*)d_in[8];
    const float* ca_se1_w   = (const float*)d_in[9];
    const float* ca_se1_b   = (const float*)d_in[10];
    const float* ca_se2_w   = (const float*)d_in[11];
    const float* ca_se2_b   = (const float*)d_in[12];
    const float* sa_gn_g    = (const float*)d_in[13];
    const float* sa_gn_b    = (const float*)d_in[14];
    const float* sa_qkv_w   = (const float*)d_in[15];
    const float* sa_proj_w  = (const float*)d_in[16];
    const float* sa_proj_b  = (const float*)d_in[17];
    const float* res_gn_g   = (const float*)d_in[18];
    const float* res_gn_b   = (const float*)d_in[19];
    const float* res_conv_w = (const float*)d_in[20];
    const float* res_conv_b = (const float*)d_in[21];
    (void)in_sizes; (void)n_in; (void)out_size; (void)ws_size;

    char* ws = (char*)d_ws;
    size_t off = 0;
    auto alloc = [&](size_t bytes) -> void* {
        void* p = ws + off;
        off = (off + bytes + 255) & ~(size_t)255;
        return p;
    };

    // padded bf16 weights
    unsigned short* wqkv = (unsigned short*)alloc((size_t)720 * 256 * 2);
    unsigned short* wprj = (unsigned short*)alloc((size_t)240 * 256 * 2);
    unsigned short* w1r  = (unsigned short*)alloc((size_t)9 * 64  * 256 * 2);
    unsigned short* w2r  = (unsigned short*)alloc((size_t)9 * 240 * 64  * 2);
    unsigned short* wrr  = (unsigned short*)alloc((size_t)9 * 240 * 256 * 2);
    // channel-last bf16 activations
    unsigned short* xsacl  = (unsigned short*)alloc((size_t)HW * 256 * 2);
    unsigned short* act1cl = (unsigned short*)alloc((size_t)HW * 256 * 2);
    unsigned short* act2cl = (unsigned short*)alloc((size_t)HW * 64  * 2);
    unsigned short* actrcl = (unsigned short*)alloc((size_t)HW * 256 * 2);
    unsigned short* hsT    = (unsigned short*)alloc((size_t)HW * 256 * 2);
    // attention tensors
    unsigned short* qkvb = (unsigned short*)alloc((size_t)720 * HW * 2);   // d-major
    unsigned short* QT   = (unsigned short*)alloc((size_t)HEADS * HW * HDP * 2);
    unsigned short* KT   = (unsigned short*)alloc((size_t)HEADS * HW * HDP * 2);
    // f32 activations
    float* c1f    = (float*)alloc((size_t)60  * HW * 4);
    float* hf     = (float*)alloc((size_t)240 * HW * 4);
    float* saf    = (float*)alloc((size_t)240 * HW * 4);
    float* fusef  = (float*)alloc((size_t)240 * HW * 4);
    float* pooled = (float*)alloc(240 * 4);
    float* yse    = (float*)alloc(240 * 4);

    auto nb = [](int n) { return (n + 255) / 256; };
    repack_w_kernel<<<nb(720 * 256), 256, 0, stream>>>(sa_qkv_w,  wqkv, 720, 240, 256);
    repack_w_kernel<<<nb(240 * 256), 256, 0, stream>>>(sa_proj_w, wprj, 240, 240, 256);
    repack_w3_kernel<<<nb(9 * 64  * 256), 256, 0, stream>>>(ca_conv1_w, w1r, 60,  240, 64,  256);
    repack_w3_kernel<<<nb(9 * 240 * 64 ), 256, 0, stream>>>(ca_conv2_w, w2r, 240, 60,  240, 64);
    repack_w3_kernel<<<nb(9 * 240 * 256), 256, 0, stream>>>(res_conv_w, wrr, 240, 240, 240, 256);

    hipMemsetAsync(xsacl,  0, (size_t)HW * 256 * 2, stream);
    hipMemsetAsync(act1cl, 0, (size_t)HW * 256 * 2, stream);
    hipMemsetAsync(act2cl, 0, (size_t)HW * 64  * 2, stream);
    hipMemsetAsync(actrcl, 0, (size_t)HW * 256 * 2, stream);
    hipMemsetAsync(hsT,    0, (size_t)HW * 256 * 2, stream);

    gn_act_kernel<<<4, 1024, 0, stream>>>(x, sa_gn_g,  sa_gn_b,  xsacl,  240, 4, 256, 0);
    gn_act_kernel<<<4, 1024, 0, stream>>>(x, ca_gn1_g, ca_gn1_b, act1cl, 240, 4, 256, 1);

    // qkv = Wqkv * gn(x) -> bf16 d-major; then token-major repack
    gemm_wmma_kernel<<<dim3(45, 16), 256, 0, stream>>>(
        wqkv, xsacl, nullptr, nullptr, nullptr, qkvb, 720, HW, 256);
    repack_qk_kernel<<<nb(HEADS * HW * HDP), 256, 0, stream>>>(qkvb, QT, KT);

    // CAB convs
    conv3_wmma_kernel<<<dim3(4, 16), 256, 0, stream>>>(
        w1r, act1cl, ca_conv1_b, nullptr, c1f, 256, 60, 64);
    gn_act_kernel<<<4, 1024, 0, stream>>>(c1f, ca_gn2_g, ca_gn2_b, act2cl, 60, 4, 64, 1);
    conv3_wmma_kernel<<<dim3(15, 16), 256, 0, stream>>>(
        w2r, act2cl, ca_conv2_b, nullptr, hf, 64, 240, 240);

    // SE
    pool_kernel<<<240, 256, 0, stream>>>(hf, pooled);
    se_kernel<<<1, 256, 0, stream>>>(pooled, ca_se1_w, ca_se1_b, ca_se2_w, ca_se2_b, yse);

    // flash attention -> channel-last hsT
    flash_attn_kernel<<<dim3(HEADS, 32), 256, 0, stream>>>(QT, KT, qkvb, hsT);

    // sa_out = proj(hs) + bias + x
    gemm_wmma_kernel<<<dim3(15, 16), 256, 0, stream>>>(
        wprj, hsT, sa_proj_b, x, saf, nullptr, 240, HW, 256);

    // fuse = ca_out + sa_out + x
    combine_kernel<<<nb(240 * HW), 256, 0, stream>>>(hf, yse, x, saf, fusef, 240 * HW);

    // residual GN+silu, final conv (+bias, +x) -> d_out
    gn_act_kernel<<<4, 1024, 0, stream>>>(fusef, res_gn_g, res_gn_b, actrcl, 240, 4, 256, 1);
    conv3_wmma_kernel<<<dim3(15, 16), 256, 0, stream>>>(
        wrr, actrcl, res_conv_b, x, (float*)d_out, 256, 240, 240);
}
